// QuanvolutionHybrid_65481071401513
// MI455X (gfx1250) — matrix operand order repaired and checked by reading, not compiled, not run on hardware
//
#include <hip/hip_runtime.h>
#include <math.h>

typedef __attribute__((ext_vector_type(16))) _Float16 v16h;
typedef __attribute__((ext_vector_type(8)))  float    v8f;

struct C2 { float r, i; };
__device__ __forceinline__ C2 cmul(C2 a, C2 b){ return {a.r*b.r - a.i*b.i, a.r*b.i + a.i*b.r}; }
__device__ __forceinline__ C2 cadd(C2 a, C2 b){ return {a.r + b.r, a.i + b.i}; }

// Workspace layout (bytes):
//   [0,1024)    : Br  f16[32 lanes][16 halves]  -> B operand for Re(psi): [G_r ; -G_i] (32x16)
//   [1024,2048) : Bi  f16[32 lanes][16 halves]  -> B operand for Im(psi): [G_i ;  G_r] (32x16)
//   [2048,2080) : Mfc float[8] = composed 1-qubit 2x2 complex matrix (r,i interleaved)
//
// f16 B-matrix (32x16, K x N) wave32 layout used here (contiguous K, same order as A):
// lane L holds column N = L&15; half h in 0..15 holds
// K = ((L<16)?0:16) + h.

__global__ void qh_setup_kernel(const float* __restrict__ wq,
                                const float* __restrict__ wfc,
                                char* __restrict__ ws)
{
    _Float16* Br  = (_Float16*)ws;
    _Float16* Bi  = (_Float16*)(ws + 1024);
    float*    Mfc = (float*)(ws + 2048);

    const int t = threadIdx.x;
    if (t < 16) {
        // Evolve basis column e_t through the shared 4-qubit gate sequence -> st[i] = G[i][t]
        C2 st[16];
        #pragma unroll
        for (int i = 0; i < 16; ++i) st[i] = { (i == t) ? 1.0f : 0.0f, 0.0f };

        for (int app = 0; app < 2; ++app) {
            for (int l = 0; l < 2; ++l) {
                for (int w = 0; w < 4; ++w) {
                    const float* g = wq + (((app * 2) + l) * 4 + w) * 3;
                    const float phi = g[0], th = g[1], om = g[2];
                    const float ct = cosf(0.5f * th),  stt  = sinf(0.5f * th);
                    const float capo = cosf(0.5f * (phi + om)), sapo = sinf(0.5f * (phi + om));
                    const float camo = cosf(0.5f * (phi - om)), samo = sinf(0.5f * (phi - om));
                    // PennyLane Rot: m00=e^{-i(phi+om)/2}ct, m01=-e^{i(phi-om)/2}st,
                    //                m10=e^{-i(phi-om)/2}st, m11=e^{i(phi+om)/2}ct
                    const C2 m00 = {  capo * ct, -sapo * ct  };
                    const C2 m01 = { -camo * stt, -samo * stt };
                    const C2 m10 = {  camo * stt, -samo * stt };
                    const C2 m11 = {  capo * ct,  sapo * ct  };
                    const int mask = 8 >> w;               // wire w <-> bit (3-w)
                    #pragma unroll
                    for (int idx = 0; idx < 16; ++idx) {
                        if (idx & mask) continue;
                        C2 a0 = st[idx], a1 = st[idx | mask];
                        st[idx]        = cadd(cmul(m00, a0), cmul(m01, a1));
                        st[idx | mask] = cadd(cmul(m10, a0), cmul(m11, a1));
                    }
                }
                const int r = 1 + l;                        // SEL ranges = (1,2)
                for (int w = 0; w < 4; ++w) {
                    const int cm = 8 >> w;
                    const int tm = 8 >> ((w + r) & 3);
                    #pragma unroll
                    for (int idx = 0; idx < 16; ++idx) {
                        if ((idx & cm) && !(idx & tm)) {    // control=1, target=0: swap pair
                            C2 tmp = st[idx]; st[idx] = st[idx | tm]; st[idx | tm] = tmp;
                        }
                    }
                }
            }
        }
        // Write B operands. Thread t owns K in {t, 16+t}.
        #pragma unroll
        for (int i = 0; i < 16; ++i) {
            Br[i * 16 + t]        = (_Float16)( st[i].r);   // lane i,     K=t     :  G_r[i][t]
            Bi[i * 16 + t]        = (_Float16)( st[i].i);   //                     :  G_i[i][t]
            Br[(16 + i) * 16 + t] = (_Float16)(-st[i].i);   // lane 16+i,  K=16+t  : -G_i[i][t]
            Bi[(16 + i) * 16 + t] = (_Float16)( st[i].r);   //                     :  G_r[i][t]
        }
    }

    if (t == 0) {
        // Compose the 9 shared Rot gates of the 1-qubit head: M = U_last * ... * U_first
        C2 M00 = {1,0}, M01 = {0,0}, M10 = {0,0}, M11 = {1,0};
        for (int app = 0; app < 3; ++app) {
            for (int l = 0; l < 3; ++l) {
                const float* g = wfc + (app * 3 + l) * 3;
                const float phi = g[0], th = g[1], om = g[2];
                const float ct = cosf(0.5f * th),  stt  = sinf(0.5f * th);
                const float capo = cosf(0.5f * (phi + om)), sapo = sinf(0.5f * (phi + om));
                const float camo = cosf(0.5f * (phi - om)), samo = sinf(0.5f * (phi - om));
                const C2 u00 = {  capo * ct, -sapo * ct  };
                const C2 u01 = { -camo * stt, -samo * stt };
                const C2 u10 = {  camo * stt, -samo * stt };
                const C2 u11 = {  capo * ct,  sapo * ct  };
                C2 n00 = cadd(cmul(u00, M00), cmul(u01, M10));
                C2 n01 = cadd(cmul(u00, M01), cmul(u01, M11));
                C2 n10 = cadd(cmul(u10, M00), cmul(u11, M10));
                C2 n11 = cadd(cmul(u10, M01), cmul(u11, M11));
                M00 = n00; M01 = n01; M10 = n10; M11 = n11;
            }
        }
        Mfc[0] = M00.r; Mfc[1] = M00.i; Mfc[2] = M01.r; Mfc[3] = M01.i;
        Mfc[4] = M10.r; Mfc[5] = M10.i; Mfc[6] = M11.r; Mfc[7] = M11.i;
    }
}

// One wave per 16-sample tile. 2x v_wmma_f32_16x16x32_f16 per tile computes the
// batched complex matvec psi = G * init.
__global__ void qh_main_kernel(const float* __restrict__ x,
                               const float* __restrict__ Wout,
                               const float* __restrict__ bout,
                               const char* __restrict__ ws,
                               float* __restrict__ out,
                               int nTiles)
{
    const int lane = threadIdx.x & 31;
    const int wave = threadIdx.x >> 5;
    const int tile = blockIdx.x * (blockDim.x >> 5) + wave;
    if (tile >= nTiles) return;                 // wave-uniform: EXEC stays all-1 for WMMA

    const int grp = lane >> 4;                  // lane group (K split / C-row split)
    const int i16 = lane & 15;

    // ---- init product state for sample m = tile*16 + i16 (lanes L and L+16 share m)
    const float* xp = x + (size_t)(tile * 16 + i16) * 784;
    const float ang0 = xp[0], ang1 = xp[1], ang2 = xp[28], ang3 = xp[29];
    float cw[4], sw[4];
    cw[0] = cosf(0.5f * ang0); sw[0] = sinf(0.5f * ang0);
    cw[1] = cosf(0.5f * ang1); sw[1] = sinf(0.5f * ang1);
    cw[2] = cosf(0.5f * ang2); sw[2] = sinf(0.5f * ang2);
    cw[3] = cosf(0.5f * ang3); sw[3] = sinf(0.5f * ang3);

    float inr[16], ini[16];
    #pragma unroll
    for (int j = 0; j < 16; ++j) {
        const float mag = ((j & 8) ? sw[0] : cw[0]) * ((j & 4) ? sw[1] : cw[1]) *
                          ((j & 2) ? sw[2] : cw[2]) * ((j & 1) ? sw[3] : cw[3]);
        const int p = __popc(j) & 3;            // phase (-i)^popcount
        inr[j] = (p == 0) ? mag : ((p == 2) ? -mag : 0.0f);
        ini[j] = (p == 1) ? -mag : ((p == 3) ? mag : 0.0f);
    }

    // ---- A operand (16x32 f16, contiguous k = 16*grp + h): A[m, k<16]=init_r[k], A[m, k>=16]=init_i[k-16]
    v16h av;
    #pragma unroll
    for (int h = 0; h < 16; ++h) {
        const int k = (grp == 0) ? h                           // K: 0..15
                                 : h + 16;                     // K: 16..31
        const float v = (k < 16) ? inr[k] : ini[k - 16];
        av[h] = (_Float16)v;
    }

    const v16h* Brv = (const v16h*)ws;
    const v16h* Biv = (const v16h*)(ws + 1024);
    const v16h br = Brv[lane];
    const v16h bi = Biv[lane];

    v8f zc = {0.f, 0.f, 0.f, 0.f, 0.f, 0.f, 0.f, 0.f};
    // Re(psi)[m,i] and Im(psi)[m,i] for the 16-sample tile
    v8f pr = __builtin_amdgcn_wmma_f32_16x16x32_f16(false, av, false, br, (short)0, zc, false, false);
    v8f pim = __builtin_amdgcn_wmma_f32_16x16x32_f16(false, av, false, bi, (short)0, zc, false, false);

    // C/D layout: lane holds column i = i16, rows m = v + 8*grp.
    // feat[m] = sum_{i<8} |psi|^2 - sum_{i>=8} |psi|^2
    const float sgn = (i16 < 8) ? 1.0f : -1.0f;
    float sarr[8];
    #pragma unroll
    for (int v = 0; v < 8; ++v) sarr[v] = sgn * (pr[v] * pr[v] + pim[v] * pim[v]);
    #pragma unroll
    for (int m = 1; m < 16; m <<= 1) {
        #pragma unroll
        for (int v = 0; v < 8; ++v) sarr[v] += __shfl_xor(sarr[v], m, 32);
    }
    // now lane (group grp) holds feats of samples m = 8*grp + v, v=0..7

    if (i16 < 8) {
        const int    mloc = grp * 8 + i16;
        const size_t s    = (size_t)tile * 16 + mloc;
        const float  f    = sarr[i16];

        const float* M = (const float*)(ws + 2048);
        const float cf = cosf(0.5f * f), sf = sinf(0.5f * f);
        // state after RX(f): a=(cf,0), b=(0,-sf);  s = Mfc * [a;b]
        const float s0r = M[0] * cf + M[3] * sf;
        const float s0i = M[1] * cf - M[2] * sf;
        const float s1r = M[4] * cf + M[7] * sf;
        const float s1i = M[5] * cf - M[6] * sf;
        const float q = (s0r * s0r + s0i * s0i) - (s1r * s1r + s1i * s1i);

        float lg[10];
        float mx = -1e30f;
        #pragma unroll
        for (int j = 0; j < 10; ++j) { lg[j] = q * Wout[j] + bout[j]; mx = fmaxf(mx, lg[j]); }
        float sum = 0.0f;
        #pragma unroll
        for (int j = 0; j < 10; ++j) sum += expf(lg[j] - mx);
        const float lse = mx + logf(sum);
        float* o = out + s * 10;
        #pragma unroll
        for (int j = 0; j < 10; ++j) o[j] = lg[j] - lse;
    }
}

extern "C" void kernel_launch(void* const* d_in, const int* in_sizes, int n_in,
                              void* d_out, int out_size, void* d_ws, size_t ws_size,
                              hipStream_t stream)
{
    const float* x   = (const float*)d_in[0];   // [B,784]
    const float* wq  = (const float*)d_in[1];   // [2,2,4,3]
    const float* wfc = (const float*)d_in[2];   // [3,3,1,3]
    const float* Wo  = (const float*)d_in[3];   // [10,1]
    const float* bo  = (const float*)d_in[4];   // [10]
    float* out = (float*)d_out;                 // [B,10] log-probs

    const int B      = in_sizes[0] / 784;       // 4096
    const int nTiles = (B + 15) / 16;           // 256 (one wave each)

    qh_setup_kernel<<<1, 32, 0, stream>>>(wq, wfc, (char*)d_ws);

    const int wavesPerBlock = 8;                // 256 threads = 8 wave32 per WGP launch
    const int blocks = (nTiles + wavesPerBlock - 1) / wavesPerBlock;
    qh_main_kernel<<<blocks, wavesPerBlock * 32, 0, stream>>>(
        x, Wo, bo, (const char*)d_ws, out, nTiles);
}
